// SparseAttnBottleneck_19688130085651
// MI455X (gfx1250) — matrix operand order repaired and checked by reading, not compiled, not run on hardware
//
#include <hip/hip_runtime.h>
#include <hip/hip_bf16.h>
#include <stdint.h>

#define DIM   512
#define VOC   32768
#define BATCH 4096
#define TOPK  64

typedef unsigned int u32;
typedef unsigned short u16;
typedef __attribute__((ext_vector_type(16))) __bf16 v16bf;
typedef __attribute__((ext_vector_type(8)))  float  v8f;
typedef __attribute__((ext_vector_type(4)))  float  f4;
typedef __attribute__((ext_vector_type(4)))  u32    u32x4;
typedef __attribute__((ext_vector_type(8)))  int    i32x8;
typedef __attribute__((ext_vector_type(4)))  int    i32x4;

union Frag { v16bf v; f4 q[2]; };

__device__ __forceinline__ u16 f2bf(float f) {
    u32 u = __float_as_uint(f);
    u32 r = (u + 0x7fffu + ((u >> 16) & 1u)) >> 16;   // RNE
    return (u16)r;
}

// monotone float<->uint key mapping for radix select
__device__ __forceinline__ u32 fmap(float f) {
    u32 u = __float_as_uint(f);
    return (u & 0x80000000u) ? ~u : (u | 0x80000000u);
}
__device__ __forceinline__ float funmap(u32 k) {
    u32 u = (k & 0x80000000u) ? (k ^ 0x80000000u) : ~k;
    return __uint_as_float(u);
}

#if __has_builtin(__builtin_amdgcn_tensor_load_to_lds)
#define HAVE_TDM 1
#else
#define HAVE_TDM 0
#endif

#if HAVE_TDM
// TDM 2-D tile load, bf16 elements: tile_d1 rows x tile_d0 elems, global row
// stride = stride0 elems, into LDS at byte offset lds_off. Optional LDS row
// padding: after every (8<<pad_int) bytes insert (pad_amt+1) DWORDs.
// D# layout per CDNA5 ISA Tensor Data Mover spec (group0 / group1).
__device__ __forceinline__ void tdm_load_tile_bf16(
    unsigned lds_off, const void* gsrc,
    unsigned tensor_d0, unsigned tensor_d1,
    unsigned tile_d0, unsigned tile_d1, unsigned stride0,
    unsigned pad_en, unsigned pad_int, unsigned pad_amt)
{
    unsigned long long ga = (unsigned long long)(uintptr_t)gsrc;
    u32x4 g0;
    g0.x = 1u;                                                     // count=1 (valid D#)
    g0.y = (u32)__builtin_amdgcn_readfirstlane((int)lds_off);      // lds_addr
    g0.z = (u32)__builtin_amdgcn_readfirstlane((int)(u32)ga);      // global_addr[31:0]
    g0.w = ((u32)__builtin_amdgcn_readfirstlane((int)(u32)(ga >> 32)) & 0x01FFFFFFu)
           | 0x80000000u;                                          // addr[56:32] | type=2
    i32x8 g1;
    g1[0] = (int)((1u << 16)                                       // data_size = 2 bytes
                  | (pad_en << 20) | (pad_int << 22) | (pad_amt << 25));
    g1[1] = (int)((tensor_d0 & 0xFFFFu) << 16);                    // tensor_dim0 lo16
    g1[2] = (int)(((tensor_d0 >> 16) & 0xFFFFu) | ((tensor_d1 & 0xFFFFu) << 16));
    g1[3] = (int)(((tensor_d1 >> 16) & 0xFFFFu) | ((tile_d0 & 0xFFFFu) << 16));
    g1[4] = (int)(tile_d1 & 0xFFFFu);                              // tile_dim1, tile_dim2=0
    g1[5] = (int)stride0;                                          // tensor_dim0_stride lo32
    g1[6] = 0;
    g1[7] = 0;
    i32x4 z = {0, 0, 0, 0};
#if __clang_major__ >= 23
    i32x8 z8 = {0, 0, 0, 0, 0, 0, 0, 0};
    __builtin_amdgcn_tensor_load_to_lds(g0, g1, z, z, z8, 0);
#else
    __builtin_amdgcn_tensor_load_to_lds(g0, g1, z, z, 0);
#endif
}
#endif

// ---------------------------------------------------------------------------
// Kernel 1: C[M x 512] = A[M x 512] @ W[512 x 512] + bias, via bf16 WMMA.
// Grid: (M/64, 512/128). 256 threads = 8 waves; wave w owns n-subtile w,
// loops 4 m-subtiles. Output bf16 (Q,K) or f32 (V).
// Tile rows padded to 40 u16 (80 B) -> 20-bank rotation, conflict-free frags.
// ---------------------------------------------------------------------------
__global__ __launch_bounds__(256) void proj_gemm(
    const float* __restrict__ A, const float* __restrict__ W,
    const float* __restrict__ bias, void* __restrict__ Cv, int bf16out)
{
    __shared__ u16 As[64][40];     // A chunk, bf16 (rows padded)
    __shared__ u16 Wt[128][40];    // W chunk transposed: [n][k] (rows padded)

    const int tid  = threadIdx.x;
    const int lane = tid & 31;
    const int wave = tid >> 5;
    const int m0 = blockIdx.x * 64;
    const int n0 = blockIdx.y * 128;

    v8f acc[4];
    const v8f vzero = {0.f,0.f,0.f,0.f,0.f,0.f,0.f,0.f};
#pragma unroll
    for (int i = 0; i < 4; ++i) acc[i] = vzero;

    for (int k0 = 0; k0 < DIM; k0 += 32) {
        __syncthreads();
        // stage A chunk (64x32 f32 -> bf16)
#pragma unroll
        for (int i = 0; i < 8; ++i) {
            int idx = tid * 8 + i;
            int r = idx >> 5, c = idx & 31;
            As[r][c] = f2bf(A[(size_t)(m0 + r) * DIM + k0 + c]);
        }
        // stage W chunk transposed (32x128 -> Wt[128][32])
        {
            int n = tid & 127;
            int kb = (tid >> 7) * 16;
#pragma unroll
            for (int i = 0; i < 16; ++i) {
                int kk = kb + i;
                Wt[n][kk] = f2bf(W[(size_t)(k0 + kk) * DIM + n0 + n]);
            }
        }
        __syncthreads();

        // B fragment: lane col n=lane%16, K = (lane/16)*8..+7 and 16+(lane/16)*8..+7
        Frag bf_;
        {
            int bn = wave * 16 + (lane & 15);
            int bk = (lane >> 4) * 8;
            bf_.q[0] = *(const f4*)&Wt[bn][bk];
            bf_.q[1] = *(const f4*)&Wt[bn][bk + 16];
        }
#pragma unroll
        for (int ms = 0; ms < 4; ++ms) {
            Frag af;
            int ar = ms * 16 + (lane & 15);
            int ak = (lane >> 4) * 8;
            af.q[0] = *(const f4*)&As[ar][ak];
            af.q[1] = *(const f4*)&As[ar][ak + 16];
            acc[ms] = __builtin_amdgcn_wmma_f32_16x16x32_bf16(
                false, af.v, false, bf_.v, (short)0, acc[ms], false, false);
        }
    }

    // bias + store (C layout: n = lane%16, m = vgpr + 8*(lane/16))
    int n = n0 + wave * 16 + (lane & 15);
    float bv = bias[n];
#pragma unroll
    for (int ms = 0; ms < 4; ++ms) {
#pragma unroll
        for (int r = 0; r < 8; ++r) {
            int m = m0 + ms * 16 + r + ((lane >> 4) << 3);
            float val = acc[ms][r] + bv;
            if (bf16out) ((u16*)Cv)[(size_t)m * DIM + n] = f2bf(val);
            else         ((float*)Cv)[(size_t)m * DIM + n] = val;
        }
    }
}

// ---------------------------------------------------------------------------
// Kernel 2: dots[BATCH x VOC] = Q(bf16) @ K(bf16)^T, f32 accumulate.
// Grid: (BATCH/64, VOC/64). 64x64 S block, 8 waves x 2 16x16 subtiles.
// 4 stages of depth-128; each stage: per-wave TDM DMA of 8 rows x 128 elems
// (2 KB) for Q and K tiles (LDS rows padded by the DMA engine itself via
// D# pad fields), then 4 k-chunks x 2 WMMAs -> 8 WMMAs per barrier pair.
// ---------------------------------------------------------------------------
#define DK   128                 // staged K-depth per stage
#define DKP  (DK + 8)            // padded LDS row length (272 B -> 4-bank rot.)

__global__ __launch_bounds__(256) void dots_kernel(
    const u16* __restrict__ Q, const u16* __restrict__ K, float* __restrict__ S)
{
    __shared__ u16 Qs[64][DKP];
    __shared__ u16 Ks[64][DKP];

    const int tid  = threadIdx.x;
    const int lane = tid & 31;
    const int wave = tid >> 5;
    const int m0 = blockIdx.x * 64;
    const int n0 = blockIdx.y * 64;

    const int ms  = wave >> 1;          // subtiles 2w,2w+1 share the m-subtile
    const int ns0 = (2 * wave) & 3;
    const int ns1 = ns0 + 1;

    const v8f vzero = {0.f,0.f,0.f,0.f,0.f,0.f,0.f,0.f};
    v8f acc0 = vzero, acc1 = vzero;

    for (int k0 = 0; k0 < DIM; k0 += DK) {
        __syncthreads();
#if HAVE_TDM
        // pad: row = 128 bf16 = 64 DWORDs (code 5); pad 4 DWORDs (code 3)
        tdm_load_tile_bf16((unsigned)(uintptr_t)&Qs[wave * 8][0],
                           Q + (size_t)(m0 + wave * 8) * DIM + k0,
                           DIM, BATCH, DK, 8, DIM, 1u, 5u, 3u);
        tdm_load_tile_bf16((unsigned)(uintptr_t)&Ks[wave * 8][0],
                           K + (size_t)(n0 + wave * 8) * DIM + k0,
                           DIM, VOC, DK, 8, DIM, 1u, 5u, 3u);
        __builtin_amdgcn_s_wait_tensorcnt(0);
#else
        { // fallback: each lane copies 16B x (DK/32) per tile
            int r = lane >> 2, cb = (lane & 3) * 8;
#pragma unroll
            for (int c = 0; c < DK; c += 32) {
                *(f4*)&Qs[wave * 8 + r][c + cb] =
                    *(const f4*)&Q[(size_t)(m0 + wave * 8 + r) * DIM + k0 + c + cb];
                *(f4*)&Ks[wave * 8 + r][c + cb] =
                    *(const f4*)&K[(size_t)(n0 + wave * 8 + r) * DIM + k0 + c + cb];
            }
        }
#endif
        __syncthreads();

#pragma unroll
        for (int kk = 0; kk < DK; kk += 32) {
            Frag af, b0, b1;
            int ar = ms * 16 + (lane & 15);
            int ak = kk + (lane >> 4) * 8;
            af.q[0] = *(const f4*)&Qs[ar][ak];
            af.q[1] = *(const f4*)&Qs[ar][ak + 16];
            int bk = kk + (lane >> 4) * 8;
            int br0 = ns0 * 16 + (lane & 15);
            int br1 = ns1 * 16 + (lane & 15);
            b0.q[0] = *(const f4*)&Ks[br0][bk];
            b0.q[1] = *(const f4*)&Ks[br0][bk + 16];
            b1.q[0] = *(const f4*)&Ks[br1][bk];
            b1.q[1] = *(const f4*)&Ks[br1][bk + 16];
            acc0 = __builtin_amdgcn_wmma_f32_16x16x32_bf16(
                false, af.v, false, b0.v, (short)0, acc0, false, false);
            acc1 = __builtin_amdgcn_wmma_f32_16x16x32_bf16(
                false, af.v, false, b1.v, (short)0, acc1, false, false);
        }
    }

#pragma unroll
    for (int r = 0; r < 8; ++r) {
        int m = m0 + ms * 16 + r + ((lane >> 4) << 3);
        S[(size_t)m * VOC + n0 + ns0 * 16 + (lane & 15)] = acc0[r];
        S[(size_t)m * VOC + n0 + ns1 * 16 + (lane & 15)] = acc1[r];
    }
}

// ---------------------------------------------------------------------------
// Kernel 3: per row: exact 64th-largest via 3-level radix histogram
// (12+12+8 bits), then masked softmax fused with out = attn @ V.
// Grid: BATCH blocks x 256 threads. Deterministic accumulation order.
// ---------------------------------------------------------------------------
__global__ __launch_bounds__(256) void topk_softmax_out(
    const float* __restrict__ S, const float* __restrict__ V,
    float* __restrict__ out)
{
    __shared__ u32 hist[4096];
    __shared__ u32 s_target, s_prefix, s_keymax;
    __shared__ float wls[256];

    const int tid = threadIdx.x;
    const size_t row = blockIdx.x;
    const float* rp = S + row * (size_t)VOC;

    for (int b = tid; b < 4096; b += 256) hist[b] = 0u;
    if (tid == 0) { s_keymax = 0u; s_target = TOPK; }
    __syncthreads();

    // level 1: top-12-bit histogram + row max
    u32 lmax = 0u;
    for (int i = 0; i < VOC / 256; ++i) {
        if (i + 8 < VOC / 256) __builtin_prefetch(&rp[tid + 256 * (i + 8)], 0, 1);
        u32 k = fmap(rp[tid + 256 * i]);
        lmax = k > lmax ? k : lmax;
        atomicAdd(&hist[k >> 20], 1u);
    }
    atomicMax(&s_keymax, lmax);
    __syncthreads();
    if (tid == 0) {
        u32 c = s_target;
        for (int b = 4095; b >= 0; --b) {
            u32 h = hist[b];
            if (h >= c) { s_prefix = (u32)b; s_target = c; break; }
            c -= h;
        }
    }
    __syncthreads();
    u32 p1 = s_prefix;
    for (int b = tid; b < 4096; b += 256) hist[b] = 0u;
    __syncthreads();

    // level 2: next 12 bits, restricted to bin p1
    for (int i = 0; i < VOC / 256; ++i) {
        u32 k = fmap(rp[tid + 256 * i]);
        if ((k >> 20) == p1) atomicAdd(&hist[(k >> 8) & 0xFFFu], 1u);
    }
    __syncthreads();
    if (tid == 0) {
        u32 c = s_target;
        for (int b = 4095; b >= 0; --b) {
            u32 h = hist[b];
            if (h >= c) { s_prefix = (p1 << 12) | (u32)b; s_target = c; break; }
            c -= h;
        }
    }
    __syncthreads();
    u32 p2 = s_prefix;
    for (int b = tid; b < 256; b += 256) hist[b] = 0u;
    __syncthreads();

    // level 3: low 8 bits -> exact 64th-largest key
    for (int i = 0; i < VOC / 256; ++i) {
        u32 k = fmap(rp[tid + 256 * i]);
        if ((k >> 8) == p2) atomicAdd(&hist[k & 0xFFu], 1u);
    }
    __syncthreads();
    if (tid == 0) {
        u32 c = s_target;
        for (int b = 255; b >= 0; --b) {
            u32 h = hist[b];
            if (h >= c) { s_prefix = (p2 << 8) | (u32)b; break; }
            c -= h;
        }
    }
    __syncthreads();
    const u32 kth = s_prefix;                 // key of 64th-largest value
    const float m = funmap(s_keymax);         // row max for stable softmax

    // fused masked softmax + attn @ V, deterministic order
    float r0 = 0.f, r1 = 0.f, ssum = 0.f;
    for (int jb = 0; jb < VOC; jb += 256) {
        __syncthreads();
        {
            if (jb + 256 < VOC) __builtin_prefetch(&rp[jb + 256 + tid], 0, 1);
            float x = rp[jb + tid];
            wls[tid] = (fmap(x) >= kth) ? __expf(x - m) : 0.0f;
        }
        __syncthreads();
        for (int jj = 0; jj < 256; ++jj) {
            float w = wls[jj];
            if (w != 0.0f) {
                ssum += w;
                const float* vp = V + (size_t)(jb + jj) * DIM;
                r0 += w * vp[tid];
                r1 += w * vp[tid + 256];
            }
        }
    }
    float inv = 1.0f / ssum;
    out[row * DIM + tid]       = r0 * inv;
    out[row * DIM + tid + 256] = r1 * inv;
}

// ---------------------------------------------------------------------------
extern "C" void kernel_launch(void* const* d_in, const int* in_sizes, int n_in,
                              void* d_out, int out_size, void* d_ws, size_t ws_size,
                              hipStream_t stream) {
    const float* x        = (const float*)d_in[0];
    const float* codebook = (const float*)d_in[1];
    const float* Wq       = (const float*)d_in[2];
    const float* bq       = (const float*)d_in[3];
    const float* Wk       = (const float*)d_in[4];
    const float* bk       = (const float*)d_in[5];
    const float* Wv       = (const float*)d_in[6];
    const float* bv       = (const float*)d_in[7];
    float* out = (float*)d_out;

    // workspace layout: Q bf16 (4 MB) | K bf16 (32 MB) | V f32 (64 MB) | dots f32 (512 MB)
    u16*   Qb = (u16*)d_ws;
    u16*   Kb = Qb + (size_t)BATCH * DIM;
    float* Vf = (float*)(Kb + (size_t)VOC * DIM);
    float* Sd = Vf + (size_t)VOC * DIM;

    proj_gemm<<<dim3(BATCH / 64, DIM / 128), 256, 0, stream>>>(x, Wq, bq, Qb, 1);
    proj_gemm<<<dim3(VOC / 64,   DIM / 128), 256, 0, stream>>>(codebook, Wk, bk, Kb, 1);
    proj_gemm<<<dim3(VOC / 64,   DIM / 128), 256, 0, stream>>>(codebook, Wv, bv, Vf, 0);
    dots_kernel<<<dim3(BATCH / 64, VOC / 64), 256, 0, stream>>>(Qb, Kb, Sd);
    topk_softmax_out<<<dim3(BATCH), 256, 0, stream>>>(Sd, Vf, out);
}
